// SemantiMambacDv0_56916906606671
// MI455X (gfx1250) — matrix operand order repaired and checked by reading, not compiled, not run on hardware
//
#include <hip/hip_runtime.h>
#include <hip/hip_bf16.h>

// ---------------------------------------------------------------------------
// MI455X (gfx1250) implementation of the SS2D/VMamba fusion pipeline.
// Dense matmuls -> v_wmma_f32_16x16x32_bf16 (bf16 A/B, f32 accum), with the
// shared W panel staged in LDS per block (via the Tensor Data Mover when the
// toolchain exposes it, else a cooperative copy).  Scan is VALU (v_exp_f32)
// with LDS staging of the shared B/C vectors.
// ---------------------------------------------------------------------------

typedef __bf16 bf16_t;
typedef __attribute__((ext_vector_type(16))) __bf16 v16bf;
typedef __attribute__((ext_vector_type(8)))  __bf16 v8bf;
typedef __attribute__((ext_vector_type(8)))  float  v8f;
typedef __attribute__((ext_vector_type(4)))  unsigned int u32x4;
typedef __attribute__((ext_vector_type(8)))  int i32x8;
typedef __attribute__((ext_vector_type(4)))  int i32x4;

// Tensor Data Mover availability / arity detection (compile-safe fallback).
#if defined(__has_builtin)
#  if __has_builtin(__builtin_amdgcn_tensor_load_to_lds) && __has_builtin(__builtin_amdgcn_s_wait_tensorcnt)
#    if __has_include(<hip/amd_detail/amd_gfx1250_TDM.h>)
#      define CDNA5_TDM_ARITY 6
#    else
#      define CDNA5_TDM_ARITY 5
#    endif
#  endif
#endif

static inline unsigned int cdiv_sz(size_t a, size_t b) { return (unsigned int)((a + b - 1) / b); }

__device__ __forceinline__ float sigmoidf_(float x) { return 1.f / (1.f + __expf(-x)); }
__device__ __forceinline__ float siluf_(float x)    { return x * sigmoidf_(x); }
__device__ __forceinline__ float softplusf_(float x){ return x > 20.f ? x : log1pf(__expf(x)); }

// ---------------------------------------------------------------------------
// WMMA GEMM:  C[M,N] = A[M,K] * W[N,K]^T   (A,W bf16 row-major, f32 accum)
// K multiple of 32 (dt_rank padded host-side).  8 waves/block: 128M x 16N
// macro tile; the 16xK W panel is staged once in LDS and shared by all waves.
// LDS layout: chunk kc (32 halves of row n) at halves (kc*16+n)*40  (64B data
// + 16B pad -> 80B row stride, conflict-free across lanes 0..15, 16B aligned).
// ---------------------------------------------------------------------------
enum { EPI_NONE = 0, EPI_BIAS_SP = 1, EPI_BN_RELU = 2 };

template<int EPI, bool WF, bool WB, bool NCHW>
__global__ __launch_bounds__(256)
void gemm_wmma_bf16(const bf16_t* __restrict__ A, const bf16_t* __restrict__ W,
                    float* __restrict__ Cf, bf16_t* __restrict__ Cb,
                    int M, int N, int K,
                    const float* __restrict__ e0, const float* __restrict__ e1,
                    int hw, int nch)
{
  extern __shared__ __align__(16) bf16_t sW[];   // (K/32) * 16 * 40 halves
  const int Kc   = K >> 5;
  const int tid  = threadIdx.x;
  const int wv   = tid >> 5;           // wave in block (0..7)
  const int lane = tid & 31;
  const int lr   = lane & 15;
  const int hi   = lane >> 4;          // 0 or 1
  const int n0   = blockIdx.x * 16;
  const int m0   = blockIdx.y * 128 + wv * 16;

  // ---- stage W panel (rows n0..n0+15, all K) into LDS ----
#if defined(CDNA5_TDM_ARITY)
  if (tid < 32) {                      // wave 0 drives the TDM
    unsigned ldsbase = (unsigned)(unsigned long long)(uintptr_t)sW;
    unsigned long long ga0 = (unsigned long long)(uintptr_t)(W + (size_t)n0 * K);
    unsigned tdim1 = (unsigned)(N - n0);           // rows available (OOB rows -> 0)
    for (int kc = 0; kc < Kc; ++kc) {
      unsigned long long ga = ga0 + (unsigned long long)kc * 64ull;  // +32 halves
      u32x4 g0;
      g0[0] = 1u;                                  // count=1 (valid descriptor)
      g0[1] = ldsbase + (unsigned)kc * 1280u;      // lds_addr (bytes)
      g0[2] = (unsigned)(ga & 0xffffffffull);      // global_addr[31:0]
      g0[3] = (unsigned)((ga >> 32) & 0x1ffffffull) | (2u << 30);  // addr[56:32] | type=2
      i32x8 g1;
      g1[0] = (int)((1u << 16)                     // data_size = 2 bytes
                  | (1u << 20)                     // pad_enable
                  | (3u << 22)                     // pad_interval: 16 dwords
                  | (3u << 25));                   // pad_amount: 4 dwords
      g1[1] = (int)(32u << 16);                    // tensor_dim0 = 32 (lo16 @ bits 63:48)
      g1[2] = (int)((tdim1 & 0xffffu) << 16);      // tensor_dim1 lo16 @ bits 95:80
      g1[3] = (int)(((tdim1 >> 16) & 0xffffu) | (32u << 16));  // dim1 hi | tile_dim0=32
      g1[4] = 16;                                  // tile_dim1 = 16 rows
      g1[5] = (int)(unsigned)K;                    // tensor_dim0_stride lo32
      g1[6] = 0; g1[7] = 0;
      i32x4 z4 = {0, 0, 0, 0};
#if CDNA5_TDM_ARITY == 6
      i32x8 z8 = {0, 0, 0, 0, 0, 0, 0, 0};
      __builtin_amdgcn_tensor_load_to_lds(g0, g1, z4, z4, z8, 0);
#else
      __builtin_amdgcn_tensor_load_to_lds(g0, g1, z4, z4, 0);
#endif
    }
    __builtin_amdgcn_s_wait_tensorcnt(0);
  }
#else
  for (int u = tid; u < Kc * 64; u += 256) {       // 16B per thread per step
    int kc = u >> 6, r = u & 63;
    int nrow = r >> 2, q = r & 3;
    int gn = n0 + nrow; if (gn >= N) gn = N - 1;   // clamp (masked at store)
    v8bf v = *(const v8bf*)(W + (size_t)gn * K + kc * 32 + q * 8);
    *(v8bf*)(sW + (size_t)(kc * 16 + nrow) * 40 + q * 8) = v;
  }
#endif
  __syncthreads();

  int mA = m0 + lr; if (mA >= M) mA = M - 1;       // clamp; masked at store
  const bf16_t* __restrict__ Ar = A + (size_t)mA * K;

  v8f acc = {0.f, 0.f, 0.f, 0.f, 0.f, 0.f, 0.f, 0.f};
  for (int kk = 0, kc = 0; kk < K; kk += 32, ++kc) {
    if (kk + 32 < K) __builtin_prefetch(Ar + kk + 32, 0, 3);
    // A 16x32 fragment: lanes0-15 K 0..15, lanes16-31 K 16..31
    v8bf a0 = *(const v8bf*)(Ar + kk + hi * 16);
    v8bf a1 = *(const v8bf*)(Ar + kk + hi * 16 + 8);
    // B 32x16 fragment from LDS: lanes0-15 K 0..15, lanes16-31 K 16..31
    const bf16_t* wp = sW + (size_t)(kc * 16 + lr) * 40 + hi * 16;
    v8bf b0 = *(const v8bf*)(wp);
    v8bf b1 = *(const v8bf*)(wp + 8);
    v16bf af, bfv;
#pragma unroll
    for (int i = 0; i < 8; ++i) { af[i] = a0[i]; af[i + 8] = a1[i]; bfv[i] = b0[i]; bfv[i + 8] = b1[i]; }
    acc = __builtin_amdgcn_wmma_f32_16x16x32_bf16(false, af, false, bfv, (short)0, acc, false, false);
  }

#pragma unroll
  for (int v = 0; v < 8; ++v) {
    int m = m0 + v + hi * 8;
    int n = n0 + lr;
    if (m < M && n < N) {
      float x = acc[v];
      if (EPI == EPI_BIAS_SP)      x = softplusf_(x + e0[n]);
      else if (EPI == EPI_BN_RELU) x = fmaxf(fmaf(x, e0[n], e1[n]), 0.f);
      size_t o;
      if (NCHW) { int b = m / hw, p = m % hw; o = ((size_t)b * nch + n) * hw + p; }
      else      o = (size_t)m * N + n;
      if (WF) Cf[o] = x;
      if (WB) Cb[o] = (bf16_t)x;
    }
  }
}

template<int EPI, bool WF, bool WB, bool NCHW>
static inline void gemm_launch(hipStream_t st, const bf16_t* A, const bf16_t* W,
                               float* Cf, bf16_t* Cb, int M, int N, int K,
                               const float* e0, const float* e1, int hw = 1, int nch = 1)
{
  dim3 grid((N + 15) / 16, (M + 127) / 128);
  size_t shbytes = (size_t)(K / 32) * 1280;        // (K/32)*16*40 halves * 2B
  gemm_wmma_bf16<EPI, WF, WB, NCHW><<<grid, dim3(256), shbytes, st>>>(A, W, Cf, Cb, M, N, K, e0, e1, hw, nch);
}

// ---------------------------------------------------------------------------
// Elementwise helpers
// ---------------------------------------------------------------------------
__global__ void cast_f32_bf16_k(const float* __restrict__ in, bf16_t* __restrict__ out, size_t n)
{
  size_t i = (size_t)blockIdx.x * blockDim.x + threadIdx.x;
  if (i < n) out[i] = (bf16_t)in[i];
}

__global__ void add_cast_bf16_k(const float* __restrict__ a, const float* __restrict__ b,
                                bf16_t* __restrict__ o, size_t n)
{
  size_t i = (size_t)blockIdx.x * blockDim.x + threadIdx.x;
  if (i < n) o[i] = (bf16_t)(a[i] + b[i]);
}

// take first R columns of [rows, ld] f32, emit [rows, 32] bf16 zero-padded
__global__ void pad32_cast_k(const float* __restrict__ in, bf16_t* __restrict__ out,
                             size_t rows, int ld, int R)
{
  size_t i = (size_t)blockIdx.x * blockDim.x + threadIdx.x;
  if (i >= rows * 32) return;
  int c = (int)(i & 31);
  size_t r = i >> 5;
  out[i] = (c < R) ? (bf16_t)in[r * (size_t)ld + c] : (bf16_t)0.f;
}

// ---------------------------------------------------------------------------
// Depthwise 3x3 conv + bias + SiLU on xo (first half of xz[b,l,2*Dn]),
// scattered into 4 scan-direction copies xs[k][b][l][d] (k outermost).
// ---------------------------------------------------------------------------
__global__ void dwconv_silu_scatter_k(const float* __restrict__ xz, const float* __restrict__ cw,
                                      const float* __restrict__ cb, bf16_t* __restrict__ xs,
                                      int Bn, int S, int Dn)
{
  size_t i = (size_t)blockIdx.x * blockDim.x + threadIdx.x;
  size_t total = (size_t)Bn * S * S * Dn;
  if (i >= total) return;
  int d = (int)(i % Dn);
  size_t t = i / Dn;
  int L = S * S;
  int l = (int)(t % L);
  int b = (int)(t / L);
  int h = l / S, w = l % S;
  float acc = cb[d];
#pragma unroll
  for (int dh = -1; dh <= 1; ++dh)
#pragma unroll
    for (int dw = -1; dw <= 1; ++dw) {
      int hh = h + dh, ww = w + dw;
      if (hh >= 0 && hh < S && ww >= 0 && ww < S)
        acc += xz[((size_t)b * L + hh * S + ww) * (size_t)(2 * Dn) + d] * cw[d * 9 + (dh + 1) * 3 + (dw + 1)];
    }
  bf16_t bv = (bf16_t)siluf_(acc);
  int lT = w * S + h;
  size_t sk = (size_t)Bn * L * Dn;
  size_t bo = (size_t)b * L * Dn;
  xs[0 * sk + bo + (size_t)l * Dn + d]            = bv;
  xs[1 * sk + bo + (size_t)lT * Dn + d]           = bv;
  xs[2 * sk + bo + (size_t)(L - 1 - l) * Dn + d]  = bv;
  xs[3 * sk + bo + (size_t)(L - 1 - lT) * Dn + d] = bv;
}

// ---------------------------------------------------------------------------
// Selective scan: thread owns (k,b,d) with 16-state recurrence over L steps.
// B_t/C_t (32 shared floats per step) staged in LDS in chunks of 8 steps.
// xs/delta/ys: [4][B][L][Dn];  xdbl: [4][B][L][Cw] with Bs at col R, Cs at R+16.
// ---------------------------------------------------------------------------
__global__ __launch_bounds__(256)
void selscan_k(const bf16_t* __restrict__ xs, const float* __restrict__ delta,
               const float* __restrict__ xdbl, const float* __restrict__ A_logs,
               const float* __restrict__ Ds, float* __restrict__ ys,
               int Bn, int L, int Dn, int Cw, int R)
{
  __shared__ float sB[8][16];
  __shared__ float sC[8][16];
  int bpb = Dn >> 8;                       // blocks per (k,b)
  int kb  = blockIdx.x / bpb;              // kb = k*Bn + b
  int d   = ((blockIdx.x % bpb) << 8) + threadIdx.x;
  int k   = kb / Bn;
  size_t rowBase = (size_t)kb * L;

  float Aa[16];
#pragma unroll
  for (int n = 0; n < 16; ++n) Aa[n] = -__expf(A_logs[((size_t)k * Dn + d) * 16 + n]);
  float Dv = Ds[(size_t)k * Dn + d];
  float h[16];
#pragma unroll
  for (int n = 0; n < 16; ++n) h[n] = 0.f;

  for (int l0 = 0; l0 < L; l0 += 8) {
    int st = threadIdx.x >> 5, c = threadIdx.x & 31;     // 8 steps x 32 cols
    float v = xdbl[(rowBase + l0 + st) * Cw + R + c];
    if (c < 16) sB[st][c] = v; else sC[st][c - 16] = v;
    if (l0 + 8 < L) {
      __builtin_prefetch(&delta[(rowBase + l0 + 8) * Dn + d], 0, 3);
      __builtin_prefetch(&xdbl[(rowBase + l0 + 8 + st) * Cw + R + c], 0, 3);
    }
    __syncthreads();
#pragma unroll
    for (int s = 0; s < 8; ++s) {
      size_t ri = (rowBase + l0 + s) * Dn + d;
      float dt = delta[ri];
      float u  = (float)xs[ri];
      float du = dt * u;
      float y  = Dv * u;
#pragma unroll
      for (int n = 0; n < 16; ++n) {
        float dA = __expf(dt * Aa[n]);
        h[n] = fmaf(dA, h[n], du * sB[s][n]);
        y    = fmaf(h[n], sC[s][n], y);
      }
      ys[ri] = y;
    }
    __syncthreads();
  }
}

// ---------------------------------------------------------------------------
// Merge 4 scan directions + LayerNorm(d) + SiLU(z) gate -> bf16 [b,l,d]
// one block per pixel
// ---------------------------------------------------------------------------
__global__ __launch_bounds__(256)
void merge_ln_gate_k(const float* __restrict__ ys, const float* __restrict__ xz,
                     const float* __restrict__ g, const float* __restrict__ be,
                     bf16_t* __restrict__ out, int Bn, int S, int Dn)
{
  __shared__ float sy[1024];
  __shared__ float rs[256];
  __shared__ float rq[256];
  int L = S * S;
  int l = blockIdx.x % L;
  int b = blockIdx.x / L;
  int h = l / S, w = l % S;
  int lT = w * S + h;
  size_t sk = (size_t)Bn * L * Dn;
  size_t bo = (size_t)b * L * Dn;
  float ls = 0.f, lq = 0.f;
  for (int d = threadIdx.x; d < Dn; d += 256) {
    float v = ys[0 * sk + bo + (size_t)l * Dn + d]
            + ys[1 * sk + bo + (size_t)lT * Dn + d]
            + ys[2 * sk + bo + (size_t)(L - 1 - l) * Dn + d]
            + ys[3 * sk + bo + (size_t)(L - 1 - lT) * Dn + d];
    sy[d] = v; ls += v; lq += v * v;
  }
  rs[threadIdx.x] = ls; rq[threadIdx.x] = lq;
  __syncthreads();
  for (int o = 128; o > 0; o >>= 1) {
    if ((int)threadIdx.x < o) { rs[threadIdx.x] += rs[threadIdx.x + o]; rq[threadIdx.x] += rq[threadIdx.x + o]; }
    __syncthreads();
  }
  float mu  = rs[0] / Dn;
  float var = rq[0] / Dn - mu * mu;
  float inv = rsqrtf(var + 1e-5f);
  for (int d = threadIdx.x; d < Dn; d += 256) {
    float v = (sy[d] - mu) * inv * g[d] + be[d];
    float z = xz[((size_t)b * L + l) * (size_t)(2 * Dn) + Dn + d];
    out[bo + (size_t)l * Dn + d] = (bf16_t)(v * siluf_(z));
  }
}

// ---------------------------------------------------------------------------
// Spatial attention: gate = sigmoid((w0*max_c + w1*mean_c)*s + b), f *= gate
// ---------------------------------------------------------------------------
__global__ void spatial_att_k(float* __restrict__ f, const float* __restrict__ w,
                              const float* __restrict__ s, const float* __restrict__ b,
                              int BP, int C)
{
  int i = blockIdx.x * blockDim.x + threadIdx.x;
  if (i >= BP) return;
  float* row = f + (size_t)i * C;
  float mx = -3.4e38f, sum = 0.f;
  for (int c = 0; c < C; ++c) { float v = row[c]; mx = fmaxf(mx, v); sum += v; }
  float a = sigmoidf_((w[0] * mx + w[1] * (sum / C)) * s[0] + b[0]);
  for (int c = 0; c < C; ++c) row[c] *= a;
}

// ---------------------------------------------------------------------------
// Dense 3x3 conv + BN + ReLU (small 8x8 spatial), pixel-major in/out
// ---------------------------------------------------------------------------
__global__ void conv3x3_bn_relu_k(const float* __restrict__ in, const float* __restrict__ w,
                                  const float* __restrict__ s, const float* __restrict__ bb,
                                  float* __restrict__ outf, bf16_t* __restrict__ outb,
                                  int Bn, int S, int Ci, int Co)
{
  size_t i = (size_t)blockIdx.x * blockDim.x + threadIdx.x;
  size_t total = (size_t)Bn * S * S * Co;
  if (i >= total) return;
  int co = (int)(i % Co);
  size_t t = i / Co;
  int L = S * S;
  int p = (int)(t % L);
  int b = (int)(t / L);
  int h = p / S, ww = p % S;
  float acc = 0.f;
  for (int dh = -1; dh <= 1; ++dh)
    for (int dw = -1; dw <= 1; ++dw) {
      int hh = h + dh, wx = ww + dw;
      if (hh < 0 || hh >= S || wx < 0 || wx >= S) continue;
      const float* px = in + ((size_t)b * L + hh * S + wx) * Ci;
      const float* wr = w + (size_t)co * Ci * 9 + (dh + 1) * 3 + (dw + 1);
      for (int ci = 0; ci < Ci; ++ci) acc += px[ci] * wr[(size_t)ci * 9];
    }
  acc = fmaxf(fmaf(acc, s[co], bb[co]), 0.f);
  size_t o = ((size_t)b * L + p) * Co + co;
  outf[o] = acc;
  outb[o] = (bf16_t)acc;
}

// ---------------------------------------------------------------------------
// Bilinear resize (align-corners, matches jnp.linspace(0, H-1, oh))
// ---------------------------------------------------------------------------
__global__ void resize_bilinear_k(const float* __restrict__ in, float* __restrict__ out,
                                  int Bn, int IH, int IW, int OH, int OW, int C)
{
  size_t i = (size_t)blockIdx.x * blockDim.x + threadIdx.x;
  size_t total = (size_t)Bn * OH * OW * C;
  if (i >= total) return;
  int c = (int)(i % C);
  size_t t = i / C;
  int p = (int)(t % ((size_t)OH * OW));
  int b = (int)(t / ((size_t)OH * OW));
  int oy = p / OW, ox = p % OW;
  float fy = (OH > 1) ? oy * (float)(IH - 1) / (float)(OH - 1) : 0.f;
  float fx = (OW > 1) ? ox * (float)(IW - 1) / (float)(OW - 1) : 0.f;
  int y0 = (int)fy; int y1 = (y0 + 1 < IH) ? y0 + 1 : IH - 1; float ty = fy - y0;
  int x0 = (int)fx; int x1 = (x0 + 1 < IW) ? x0 + 1 : IW - 1; float tx = fx - x0;
  size_t base = (size_t)b * IH * IW;
  float v00 = in[(base + (size_t)y0 * IW + x0) * C + c];
  float v01 = in[(base + (size_t)y0 * IW + x1) * C + c];
  float v10 = in[(base + (size_t)y1 * IW + x0) * C + c];
  float v11 = in[(base + (size_t)y1 * IW + x1) * C + c];
  float v0 = v00 + (v01 - v00) * tx;
  float v1 = v10 + (v11 - v10) * tx;
  out[i] = v0 + (v1 - v0) * ty;
}

// ---------------------------------------------------------------------------
// Host-side orchestration
// ---------------------------------------------------------------------------
struct Bump {
  char* base; size_t off;
  void* get(size_t n) { size_t o = (off + 255) & ~(size_t)255; off = o + n; return base + o; }
};

struct SSMP {
  const float *in_proj, *conv_w, *conv_b, *x_proj, *dt_w, *dt_b, *A_logs, *Ds, *ln_g, *ln_b, *out_proj;
};

static void run_ss2d(hipStream_t st, Bump& ar, const float* x, int Bn, int S, int dmodel,
                     const SSMP& p, float* outf, bf16_t* outb)
{
  const int di = 2 * dmodel, R = dmodel / 16, Cw = R + 32, L = S * S, M = Bn * L;

  bf16_t* xb    = (bf16_t*)ar.get((size_t)M * dmodel * sizeof(bf16_t));
  bf16_t* ipw   = (bf16_t*)ar.get((size_t)2 * di * dmodel * sizeof(bf16_t));
  float*  xz    = (float*) ar.get((size_t)M * 2 * di * sizeof(float));
  bf16_t* xs    = (bf16_t*)ar.get((size_t)4 * M * di * sizeof(bf16_t));
  bf16_t* xpw   = (bf16_t*)ar.get((size_t)4 * Cw * di * sizeof(bf16_t));
  float*  xdbl  = (float*) ar.get((size_t)4 * M * Cw * sizeof(float));
  bf16_t* dtsb  = (bf16_t*)ar.get((size_t)4 * M * 32 * sizeof(bf16_t));
  bf16_t* dtwb  = (bf16_t*)ar.get((size_t)4 * di * 32 * sizeof(bf16_t));
  float*  delta = (float*) ar.get((size_t)4 * M * di * sizeof(float));
  float*  ysb   = (float*) ar.get((size_t)4 * M * di * sizeof(float));
  bf16_t* yg    = (bf16_t*)ar.get((size_t)M * di * sizeof(bf16_t));
  bf16_t* opw   = (bf16_t*)ar.get((size_t)dmodel * di * sizeof(bf16_t));

  size_t n;
  n = (size_t)M * dmodel;
  cast_f32_bf16_k<<<cdiv_sz(n, 256), 256, 0, st>>>(x, xb, n);
  n = (size_t)2 * di * dmodel;
  cast_f32_bf16_k<<<cdiv_sz(n, 256), 256, 0, st>>>(p.in_proj, ipw, n);
  gemm_launch<EPI_NONE, true, false, false>(st, xb, ipw, xz, nullptr, M, 2 * di, dmodel, nullptr, nullptr);

  n = (size_t)M * di;
  dwconv_silu_scatter_k<<<cdiv_sz(n, 256), 256, 0, st>>>(xz, p.conv_w, p.conv_b, xs, Bn, S, di);

  n = (size_t)4 * Cw * di;
  cast_f32_bf16_k<<<cdiv_sz(n, 256), 256, 0, st>>>(p.x_proj, xpw, n);
  for (int k = 0; k < 4; ++k)
    gemm_launch<EPI_NONE, true, false, false>(st, xs + (size_t)k * M * di, xpw + (size_t)k * Cw * di,
                                              xdbl + (size_t)k * M * Cw, nullptr, M, Cw, di, nullptr, nullptr);

  n = (size_t)4 * M * 32;
  pad32_cast_k<<<cdiv_sz(n, 256), 256, 0, st>>>(xdbl, dtsb, (size_t)4 * M, Cw, R);
  n = (size_t)4 * di * 32;
  pad32_cast_k<<<cdiv_sz(n, 256), 256, 0, st>>>(p.dt_w, dtwb, (size_t)4 * di, R, R);
  for (int k = 0; k < 4; ++k)
    gemm_launch<EPI_BIAS_SP, true, false, false>(st, dtsb + (size_t)k * M * 32, dtwb + (size_t)k * di * 32,
                                                 delta + (size_t)k * M * di, nullptr, M, di, 32,
                                                 p.dt_b + (size_t)k * di, nullptr);

  selscan_k<<<4 * Bn * (di / 256), 256, 0, st>>>(xs, delta, xdbl, p.A_logs, p.Ds, ysb, Bn, L, di, Cw, R);
  merge_ln_gate_k<<<Bn * L, 256, 0, st>>>(ysb, xz, p.ln_g, p.ln_b, yg, Bn, S, di);

  n = (size_t)dmodel * di;
  cast_f32_bf16_k<<<cdiv_sz(n, 256), 256, 0, st>>>(p.out_proj, opw, n);
  if (outb)
    gemm_launch<EPI_NONE, true, true, false>(st, yg, opw, outf, outb, M, dmodel, di, nullptr, nullptr);
  else
    gemm_launch<EPI_NONE, true, false, false>(st, yg, opw, outf, nullptr, M, dmodel, di, nullptr, nullptr);
}

extern "C" void kernel_launch(void* const* d_in, const int* in_sizes, int n_in,
                              void* d_out, int out_size, void* d_ws, size_t ws_size,
                              hipStream_t stream)
{
  (void)in_sizes; (void)n_in; (void)out_size; (void)ws_size;
  // setup_inputs flattening order: x1, x2, ssm1(11), ssm2(11), convs(6x {w,s,b})
  const float* x1 = (const float*)d_in[0];
  const float* x2 = (const float*)d_in[1];
  SSMP s1 { (const float*)d_in[2],  (const float*)d_in[3],  (const float*)d_in[4],
            (const float*)d_in[5],  (const float*)d_in[6],  (const float*)d_in[7],
            (const float*)d_in[8],  (const float*)d_in[9],  (const float*)d_in[10],
            (const float*)d_in[11], (const float*)d_in[12] };
  SSMP s2 { (const float*)d_in[13], (const float*)d_in[14], (const float*)d_in[15],
            (const float*)d_in[16], (const float*)d_in[17], (const float*)d_in[18],
            (const float*)d_in[19], (const float*)d_in[20], (const float*)d_in[21],
            (const float*)d_in[22], (const float*)d_in[23] };
  const float* c11w = (const float*)d_in[24]; const float* c11s = (const float*)d_in[25]; const float* c11b = (const float*)d_in[26];
  const float* c12w = (const float*)d_in[27]; const float* c12s = (const float*)d_in[28]; const float* c12b = (const float*)d_in[29];
  const float* c13w = (const float*)d_in[30]; const float* c13s = (const float*)d_in[31]; const float* c13b = (const float*)d_in[32];
  const float* saw  = (const float*)d_in[33]; const float* sas  = (const float*)d_in[34]; const float* sab  = (const float*)d_in[35];
  const float* c22w = (const float*)d_in[36]; const float* c22s = (const float*)d_in[37]; const float* c22b = (const float*)d_in[38];
  const float* c23w = (const float*)d_in[39]; const float* c23s = (const float*)d_in[40]; const float* c23b = (const float*)d_in[41];

  const int B = 8;
  Bump ar { (char*)d_ws, 0 };

  // persistent buffers
  bf16_t* out2b = (bf16_t*)ar.get((size_t)B * 64 * 512 * sizeof(bf16_t));     // ss2d(x2) out (bf16)
  float*  f1    = (float*) ar.get((size_t)B * 4096 * 128 * sizeof(float));    // upsampled stage-1 features
  float*  out1  = (float*) ar.get((size_t)B * 4096 * 128 * sizeof(float));    // ss2d(x1) out
  size_t mark = ar.off;

  // ---- Phase 1: SS2D on x2 (8x8 spatial, d_model=512) ----
  float* out2f = (float*)ar.get((size_t)B * 64 * 512 * sizeof(float));
  run_ss2d(stream, ar, x2, B, 8, 512, s2, out2f, out2b);

  // ---- Phase 2: st1conv1 -> spatial attention -> st1conv2(3x3) -> st1conv3 -> resize 8->64 ----
  ar.off = mark;
  {
    size_t n;
    bf16_t* w1 = (bf16_t*)ar.get((size_t)128 * 512 * sizeof(bf16_t));
    n = (size_t)128 * 512;
    cast_f32_bf16_k<<<cdiv_sz(n, 256), 256, 0, stream>>>(c11w, w1, n);
    float* f = (float*)ar.get((size_t)B * 64 * 128 * sizeof(float));
    gemm_launch<EPI_BN_RELU, true, false, false>(stream, out2b, w1, f, nullptr, B * 64, 128, 512, c11s, c11b);
    spatial_att_k<<<cdiv_sz((size_t)B * 64, 256), 256, 0, stream>>>(f, saw, sas, sab, B * 64, 128);

    float*  f1a  = (float*) ar.get((size_t)B * 64 * 128 * sizeof(float));
    bf16_t* f1ab = (bf16_t*)ar.get((size_t)B * 64 * 128 * sizeof(bf16_t));
    n = (size_t)B * 64 * 128;
    conv3x3_bn_relu_k<<<cdiv_sz(n, 256), 256, 0, stream>>>(f, c12w, c12s, c12b, f1a, f1ab, B, 8, 128, 128);

    bf16_t* w3 = (bf16_t*)ar.get((size_t)128 * 128 * sizeof(bf16_t));
    n = (size_t)128 * 128;
    cast_f32_bf16_k<<<cdiv_sz(n, 256), 256, 0, stream>>>(c13w, w3, n);
    float* f1b = (float*)ar.get((size_t)B * 64 * 128 * sizeof(float));
    gemm_launch<EPI_BN_RELU, true, false, false>(stream, f1ab, w3, f1b, nullptr, B * 64, 128, 128, c13s, c13b);

    n = (size_t)B * 4096 * 128;
    resize_bilinear_k<<<cdiv_sz(n, 256), 256, 0, stream>>>(f1b, f1, B, 8, 8, 64, 64, 128);
  }

  // ---- Phase 3: SS2D on x1 (64x64 spatial, d_model=128) ----
  ar.off = mark;
  run_ss2d(stream, ar, x1, B, 64, 128, s1, out1, nullptr);

  // ---- Phase 4: add, st2conv2, resize 64->256, st2conv3 x2 (NCHW out) ----
  ar.off = mark;
  {
    size_t n = (size_t)B * 4096 * 128;
    bf16_t* sb = (bf16_t*)ar.get(n * sizeof(bf16_t));
    add_cast_bf16_k<<<cdiv_sz(n, 256), 256, 0, stream>>>(f1, out1, sb, n);

    bf16_t* w4 = (bf16_t*)ar.get((size_t)64 * 128 * sizeof(bf16_t));
    n = (size_t)64 * 128;
    cast_f32_bf16_k<<<cdiv_sz(n, 256), 256, 0, stream>>>(c22w, w4, n);
    float* g2 = (float*)ar.get((size_t)B * 4096 * 64 * sizeof(float));
    gemm_launch<EPI_BN_RELU, true, false, false>(stream, sb, w4, g2, nullptr, B * 4096, 64, 128, c22s, c22b);

    float* g3 = (float*)ar.get((size_t)B * 65536 * 64 * sizeof(float));
    n = (size_t)B * 65536 * 64;
    resize_bilinear_k<<<cdiv_sz(n, 256), 256, 0, stream>>>(g2, g3, B, 64, 64, 256, 256, 64);
    bf16_t* g3b = (bf16_t*)ar.get(n * sizeof(bf16_t));
    cast_f32_bf16_k<<<cdiv_sz(n, 256), 256, 0, stream>>>(g3, g3b, n);

    bf16_t* w5 = (bf16_t*)ar.get((size_t)64 * 64 * sizeof(bf16_t));
    size_t nw = (size_t)64 * 64;
    cast_f32_bf16_k<<<cdiv_sz(nw, 256), 256, 0, stream>>>(c23w, w5, nw);

    bf16_t* g4b = (bf16_t*)ar.get(n * sizeof(bf16_t));
    gemm_launch<EPI_BN_RELU, false, true, false>(stream, g3b, w5, nullptr, g4b, B * 65536, 64, 64, c23s, c23b);
    // second st2conv3 application, scattering straight to NCHW output
    gemm_launch<EPI_BN_RELU, true, false, true>(stream, g4b, w5, (float*)d_out, nullptr,
                                                B * 65536, 64, 64, c23s, c23b, 65536, 64);
  }
}
